// TransformerDecoder_88003879895629
// MI455X (gfx1250) — matrix operand order repaired and checked by reading, not compiled, not run on hardware
//
#include <hip/hip_runtime.h>
#include <hip/hip_bf16.h>

// ---------------- problem constants (compile-time) ----------------
#define B_  2
#define LD_ 512
#define LE_ 1024
#define H_  1024
#define FI_ 4096
#define NH_ 16
#define DH_ 64
#define NL_ 4

typedef __bf16 bf16t;
typedef __attribute__((ext_vector_type(16))) __bf16 v16bf;
typedef __attribute__((ext_vector_type(8)))  __bf16 v8bf;
typedef __attribute__((ext_vector_type(4)))  __bf16 v4bf;
typedef __attribute__((ext_vector_type(8)))  float  v8f;
typedef __attribute__((ext_vector_type(4)))  float  v4f;

// ---- CDNA5 async global->LDS path (guarded: falls back to plain loads) ----
#if __has_builtin(__builtin_amdgcn_global_load_async_to_lds_b128)
#define ASYNC_LDS 1
typedef int vint4g __attribute__((vector_size(4 * sizeof(int))));
typedef __attribute__((address_space(1))) vint4g* g_v4p;
typedef __attribute__((address_space(3))) vint4g* l_v4p;
#endif

__device__ __forceinline__ void async_wait() {
#ifdef ASYNC_LDS
#if __has_builtin(__builtin_amdgcn_s_wait_asynccnt)
  __builtin_amdgcn_s_wait_asynccnt(0);
#else
  asm volatile("s_wait_asynccnt 0x0" ::: "memory");
#endif
#endif
}

// D = A(16x32 bf16) x B(32x16 bf16) + C(f32), wave32 WMMA
__device__ __forceinline__ v8f wmma_bf16(v16bf a, v16bf b, v8f c) {
  return __builtin_amdgcn_wmma_f32_16x16x32_bf16(false, a, false, b, (short)0, c,
                                                 false, false);
}

__device__ __forceinline__ v16bf frag16(const bf16t* p0, const bf16t* p1) {
  v8bf lo = *(const v8bf*)p0;
  v8bf hi = *(const v8bf*)p1;
  return __builtin_shufflevector(lo, hi, 0,1,2,3,4,5,6,7,8,9,10,11,12,13,14,15);
}
// A fragment (16x32, MxK): lane l (m=l%16, hlf=l/16): e<8 -> k=hlf*8+e ; e>=8 -> k=16+hlf*8+(e-8)
__device__ __forceinline__ v16bf frag_a(const bf16t* row, int hlf) {
  return frag16(row + hlf * 8, row + 16 + hlf * 8);
}
// B fragment (32x16, KxN): lane l (n=l%16, hlf=l/16): e<8 -> k=hlf*8+e ; e>=8 -> k=16+hlf*8+(e-8)
__device__ __forceinline__ v16bf frag_b(const bf16t* row, int hlf) {
  return frag16(row + hlf * 8, row + 16 + hlf * 8);
}

__device__ __forceinline__ v4bf pack4(v4f v) {
  v4bf p; p.x = (bf16t)v.x; p.y = (bf16t)v.y; p.z = (bf16t)v.z; p.w = (bf16t)v.w;
  return p;
}

// stage fp32 activations -> bf16 LDS tile [128][32]
__device__ __forceinline__ void stage_a(const float* __restrict__ A, int K,
                                        int m0, int k0, int tid,
                                        bf16t (*as)[32]) {
  #pragma unroll
  for (int it = 0; it < 4; ++it) {
    int idx = tid + it * 256;
    int r = idx >> 3, c = (idx & 7) * 4;
    v4f v = *(const v4f*)(A + (size_t)(m0 + r) * K + k0 + c);
    *(v4bf*)&as[r][c] = pack4(v);
  }
}

// stage pre-transposed bf16 weights [N][K] -> LDS tile [128][32]
__device__ __forceinline__ void stage_b(const bf16t* __restrict__ Wt, int K,
                                        int n0, int k0, int tid,
                                        bf16t (*bs)[32]) {
  #pragma unroll
  for (int it = 0; it < 2; ++it) {
    int idx = tid + it * 256;
    int r = idx >> 2, c = (idx & 3) * 8;     // 16B chunks
    const bf16t* gp = Wt + (size_t)(n0 + r) * K + k0 + c;
#ifdef ASYNC_LDS
    __builtin_amdgcn_global_load_async_to_lds_b128((g_v4p)(void*)gp,
                                                   (l_v4p)(void*)&bs[r][c], 0, 0);
#else
    *(v8bf*)&bs[r][c] = *(const v8bf*)gp;
#endif
  }
}

// ------------------------------------------------------------------
// GEMM: C[M,N] = act(A[M,K] @ W[K,N] + bias (+ res)), W pre-transposed bf16.
// 256 thr (8 waves), tile 128x128, K-step 32, double-buffered LDS,
// wave grid 4x2, each wave 32x64 = 2x4 wmma tiles.
// ------------------------------------------------------------------
template<bool RES, bool RELU>
__global__ __launch_bounds__(256) void gemm_kernel(
    const float* __restrict__ A, const bf16t* __restrict__ Wt,
    const float* __restrict__ bias, const float* __restrict__ res,
    float* __restrict__ C, int M, int N, int K)
{
  __shared__ __align__(16) bf16t As[2][128][32];
  __shared__ __align__(16) bf16t Bs[2][128][32];
  const int tid = threadIdx.x, wave = tid >> 5, lane = tid & 31;
  const int hlf = lane >> 4, l16 = lane & 15;
  const int m0 = blockIdx.x * 128, n0 = blockIdx.y * 128;
  const int wm = (wave >> 1) * 32, wn = (wave & 1) * 64;

  v8f acc[2][4];
  const v8f z8 = {0.f,0.f,0.f,0.f,0.f,0.f,0.f,0.f};
  #pragma unroll
  for (int i = 0; i < 2; ++i)
    #pragma unroll
    for (int j = 0; j < 4; ++j) acc[i][j] = z8;

  stage_b(Wt, K, n0, 0, tid, Bs[0]);
  stage_a(A,  K, m0, 0, tid, As[0]);
  async_wait();
  __syncthreads();

  int s = 0;
  for (int k0 = 0; k0 < K; k0 += 32, s ^= 1) {
    if (k0 + 32 < K) {                        // prefetch next tile
      stage_b(Wt, K, n0, k0 + 32, tid, Bs[s ^ 1]);
      stage_a(A,  K, m0, k0 + 32, tid, As[s ^ 1]);
    }
    v16bf af[2], bfv[4];
    af[0] = frag_a(&As[s][wm + l16][0], hlf);
    af[1] = frag_a(&As[s][wm + 16 + l16][0], hlf);
    #pragma unroll
    for (int j = 0; j < 4; ++j) bfv[j] = frag_b(&Bs[s][wn + j*16 + l16][0], hlf);
    #pragma unroll
    for (int i = 0; i < 2; ++i)
      #pragma unroll
      for (int j = 0; j < 4; ++j) acc[i][j] = wmma_bf16(af[i], bfv[j], acc[i][j]);
    async_wait();
    __syncthreads();
  }
  #pragma unroll
  for (int i = 0; i < 2; ++i)
    #pragma unroll
    for (int j = 0; j < 4; ++j) {
      const int n = n0 + wn + j * 16 + l16;
      const float bv = bias[n];
      const size_t base = (size_t)(m0 + wm + i * 16 + hlf * 8) * N + n;
      #pragma unroll
      for (int r = 0; r < 8; ++r) {
        float v = acc[i][j][r] + bv;
        if (RES)  v += res[base + (size_t)r * N];
        if (RELU) v = fmaxf(v, 0.f);
        C[base + (size_t)r * N] = v;
      }
    }
}

// ------------------------------------------------------------------
// Weight convert + transpose: Wf[K][N] f32 -> Wt[N][K] bf16 (32x32 tiles)
// ------------------------------------------------------------------
__global__ __launch_bounds__(256) void convert_w_kernel(
    const float* __restrict__ Wf, bf16t* __restrict__ Wt, int K, int N)
{
  __shared__ float t[32][33];
  const int tid = threadIdx.x;
  const int k0 = blockIdx.x * 32, n0 = blockIdx.y * 32;
  const int r = tid >> 3, c = (tid & 7) * 4;
  v4f v = *(const v4f*)(Wf + (size_t)(k0 + r) * N + n0 + c);
  t[r][c+0] = v.x; t[r][c+1] = v.y; t[r][c+2] = v.z; t[r][c+3] = v.w;
  __syncthreads();
  v4bf o;
  o.x = (bf16t)t[c+0][r]; o.y = (bf16t)t[c+1][r];
  o.z = (bf16t)t[c+2][r]; o.w = (bf16t)t[c+3][r];
  *(v4bf*)(Wt + (size_t)(n0 + r) * K + k0 + c) = o;
}

// ------------------------------------------------------------------
// Self-attention scores: s = scale * q k^T, causal+pad mask -> -1e4
// tile 64(q) x 128(k), K = DH = 64. waves 2x4, each 32x32 (2x2 tiles)
// ------------------------------------------------------------------
__global__ __launch_bounds__(256) void self_scores_kernel(
    const float* __restrict__ q, const float* __restrict__ k,
    const float* __restrict__ dmask, float* __restrict__ s)
{
  __shared__ __align__(16) bf16t Qs[64][DH_];
  __shared__ __align__(16) bf16t Ks[128][DH_];
  const int tid = threadIdx.x, wave = tid >> 5, lane = tid & 31;
  const int hlf = lane >> 4, l16 = lane & 15;
  const int bh = blockIdx.z, b = bh / NH_, h = bh % NH_;
  const int q0 = blockIdx.x * 64, k0 = blockIdx.y * 128;
  const float* qb = q + (size_t)b * LD_ * H_ + h * DH_;
  const float* kb = k + (size_t)b * LD_ * H_ + h * DH_;
  #pragma unroll
  for (int it = 0; it < 4; ++it) {
    int idx = tid + it * 256; int r = idx >> 4, c = (idx & 15) * 4;
    *(v4bf*)&Qs[r][c] = pack4(*(const v4f*)(qb + (size_t)(q0 + r) * H_ + c));
  }
  #pragma unroll
  for (int it = 0; it < 8; ++it) {
    int idx = tid + it * 256; int r = idx >> 4, c = (idx & 15) * 4;
    *(v4bf*)&Ks[r][c] = pack4(*(const v4f*)(kb + (size_t)(k0 + r) * H_ + c));
  }
  __syncthreads();
  const int wm = (wave >> 2) * 32, wn = (wave & 3) * 32;
  v8f acc[2][2];
  const v8f z8 = {0.f,0.f,0.f,0.f,0.f,0.f,0.f,0.f};
  acc[0][0] = z8; acc[0][1] = z8; acc[1][0] = z8; acc[1][1] = z8;
  #pragma unroll
  for (int ks = 0; ks < 2; ++ks) {
    v16bf a0 = frag_a(&Qs[wm + l16][ks * 32], hlf);
    v16bf a1 = frag_a(&Qs[wm + 16 + l16][ks * 32], hlf);
    v16bf b0 = frag_b(&Ks[wn + l16][ks * 32], hlf);
    v16bf b1 = frag_b(&Ks[wn + 16 + l16][ks * 32], hlf);
    acc[0][0] = wmma_bf16(a0, b0, acc[0][0]);
    acc[0][1] = wmma_bf16(a0, b1, acc[0][1]);
    acc[1][0] = wmma_bf16(a1, b0, acc[1][0]);
    acc[1][1] = wmma_bf16(a1, b1, acc[1][1]);
  }
  float* sb = s + (size_t)bh * LD_ * LD_;
  #pragma unroll
  for (int i = 0; i < 2; ++i)
    #pragma unroll
    for (int j = 0; j < 2; ++j) {
      const int n = k0 + wn + j * 16 + l16;
      const float dm = dmask[b * LD_ + n];
      #pragma unroll
      for (int r = 0; r < 8; ++r) {
        const int m = q0 + wm + i * 16 + hlf * 8 + r;
        float v = acc[i][j][r] * 0.125f;
        if (!(n <= m && dm > 0.f)) v = -1e4f;
        sb[(size_t)m * LD_ + n] = v;
      }
    }
}

// ------------------------------------------------------------------
// Cross p: e = scale * q k^T ; p = sigmoid(e); band masks; * enc_mask
// ------------------------------------------------------------------
__global__ __launch_bounds__(256) void cross_p_kernel(
    const float* __restrict__ q, const float* __restrict__ k,
    const float* __restrict__ emask, const float* __restrict__ upper,
    const float* __restrict__ lower, float* __restrict__ p)
{
  __shared__ __align__(16) bf16t Qs[64][DH_];
  __shared__ __align__(16) bf16t Ks[128][DH_];
  const int tid = threadIdx.x, wave = tid >> 5, lane = tid & 31;
  const int hlf = lane >> 4, l16 = lane & 15;
  const int bh = blockIdx.z, b = bh / NH_, h = bh % NH_;
  const int q0 = blockIdx.x * 64, k0 = blockIdx.y * 128;
  const float* qb = q + (size_t)b * LD_ * H_ + h * DH_;
  const float* kb = k + (size_t)b * LE_ * H_ + h * DH_;
  #pragma unroll
  for (int it = 0; it < 4; ++it) {
    int idx = tid + it * 256; int r = idx >> 4, c = (idx & 15) * 4;
    *(v4bf*)&Qs[r][c] = pack4(*(const v4f*)(qb + (size_t)(q0 + r) * H_ + c));
  }
  #pragma unroll
  for (int it = 0; it < 8; ++it) {
    int idx = tid + it * 256; int r = idx >> 4, c = (idx & 15) * 4;
    *(v4bf*)&Ks[r][c] = pack4(*(const v4f*)(kb + (size_t)(k0 + r) * H_ + c));
  }
  __syncthreads();
  const int wm = (wave >> 2) * 32, wn = (wave & 3) * 32;
  v8f acc[2][2];
  const v8f z8 = {0.f,0.f,0.f,0.f,0.f,0.f,0.f,0.f};
  acc[0][0] = z8; acc[0][1] = z8; acc[1][0] = z8; acc[1][1] = z8;
  #pragma unroll
  for (int ks = 0; ks < 2; ++ks) {
    v16bf a0 = frag_a(&Qs[wm + l16][ks * 32], hlf);
    v16bf a1 = frag_a(&Qs[wm + 16 + l16][ks * 32], hlf);
    v16bf b0 = frag_b(&Ks[wn + l16][ks * 32], hlf);
    v16bf b1 = frag_b(&Ks[wn + 16 + l16][ks * 32], hlf);
    acc[0][0] = wmma_bf16(a0, b0, acc[0][0]);
    acc[0][1] = wmma_bf16(a0, b1, acc[0][1]);
    acc[1][0] = wmma_bf16(a1, b0, acc[1][0]);
    acc[1][1] = wmma_bf16(a1, b1, acc[1][1]);
  }
  #pragma unroll
  for (int i = 0; i < 2; ++i)
    #pragma unroll
    for (int j = 0; j < 2; ++j) {
      const int n = k0 + wn + j * 16 + l16;
      const float em = emask[b * LE_ + n];
      #pragma unroll
      for (int r = 0; r < 8; ++r) {
        const int m = q0 + wm + i * 16 + hlf * 8 + r;
        float e = acc[i][j][r] * 0.125f;
        float pv = 1.f / (1.f + __expf(-e));
        const size_t bmn = ((size_t)b * LD_ + m) * LE_ + n;
        if (upper[bmn] > 0.f) pv = 0.f;
        if (lower[bmn] > 0.f) pv = 1.f;
        pv *= em;
        p[((size_t)bh * LD_ + m) * LE_ + n] = pv;
      }
    }
}

// ------------------------------------------------------------------
// Context: ctx[b,m,h,:] = a[bh,m,:] @ v[b,:,h,:]  (N=64, K=LK)
// tile 128(m) x 64(n), K-step 32. 8 waves, each 16 rows x 64 cols.
// ------------------------------------------------------------------
__global__ __launch_bounds__(256) void attn_context_kernel(
    const float* __restrict__ a, const float* __restrict__ v,
    float* __restrict__ ctx, int LK)
{
  __shared__ __align__(16) bf16t As[128][32];
  __shared__ __align__(16) bf16t Vt[DH_][32];   // [n][k]
  const int tid = threadIdx.x, wave = tid >> 5, lane = tid & 31;
  const int hlf = lane >> 4, l16 = lane & 15;
  const int bh = blockIdx.y, b = bh / NH_, h = bh % NH_;
  const int m0 = blockIdx.x * 128;
  const float* ab = a + (size_t)bh * LD_ * LK;
  const float* vb = v + (size_t)b * LK * H_ + h * DH_;
  v8f acc[4];
  const v8f z8 = {0.f,0.f,0.f,0.f,0.f,0.f,0.f,0.f};
  acc[0] = z8; acc[1] = z8; acc[2] = z8; acc[3] = z8;

  for (int k0 = 0; k0 < LK; k0 += 32) {
    #pragma unroll
    for (int it = 0; it < 4; ++it) {
      int idx = tid + it * 256; int r = idx >> 3, c = (idx & 7) * 4;
      *(v4bf*)&As[r][c] = pack4(*(const v4f*)(ab + (size_t)(m0 + r) * LK + k0 + c));
    }
    #pragma unroll
    for (int it = 0; it < 2; ++it) {            // V: 32x64 -> transposed
      int idx = tid + it * 256; int r = idx >> 4, c = (idx & 15) * 4;
      v4f t = *(const v4f*)(vb + (size_t)(k0 + r) * H_ + c);
      Vt[c+0][r] = (bf16t)t.x; Vt[c+1][r] = (bf16t)t.y;
      Vt[c+2][r] = (bf16t)t.z; Vt[c+3][r] = (bf16t)t.w;
    }
    __syncthreads();
    v16bf af = frag_a(&As[wave * 16 + l16][0], hlf);
    #pragma unroll
    for (int j = 0; j < 4; ++j) {
      v16bf bfv = frag_b(&Vt[j * 16 + l16][0], hlf);
      acc[j] = wmma_bf16(af, bfv, acc[j]);
    }
    __syncthreads();
  }
  #pragma unroll
  for (int j = 0; j < 4; ++j)
    #pragma unroll
    for (int r = 0; r < 8; ++r) {
      const int m = m0 + wave * 16 + hlf * 8 + r;
      const int n = j * 16 + l16;
      ctx[((size_t)b * LD_ + m) * H_ + h * DH_ + n] = acc[j][r];
    }
}

// ------------------------------------------------------------------
__global__ __launch_bounds__(256) void softmax_kernel(float* __restrict__ s, int LK)
{
  __shared__ float red[256];
  const int tid = threadIdx.x;
  float* sr = s + (size_t)blockIdx.x * LK;
  float mx = -3.4e38f;
  for (int j = tid; j < LK; j += 256) mx = fmaxf(mx, sr[j]);
  red[tid] = mx; __syncthreads();
  for (int off = 128; off > 0; off >>= 1) {
    if (tid < off) red[tid] = fmaxf(red[tid], red[tid + off]);
    __syncthreads();
  }
  mx = red[0]; __syncthreads();
  float sum = 0.f;
  for (int j = tid; j < LK; j += 256) { float e = __expf(sr[j] - mx); sr[j] = e; sum += e; }
  red[tid] = sum; __syncthreads();
  for (int off = 128; off > 0; off >>= 1) {
    if (tid < off) red[tid] += red[tid + off];
    __syncthreads();
  }
  float inv = 1.f / red[0];
  for (int j = tid; j < LK; j += 256) sr[j] *= inv;
}

// ------------------------------------------------------------------
__global__ __launch_bounds__(256) void layernorm_kernel(
    const float* __restrict__ x, const float* __restrict__ g,
    const float* __restrict__ bta, float* __restrict__ y)
{
  __shared__ float red[256];
  const int tid = threadIdx.x;
  const float* xr = x + (size_t)blockIdx.x * H_;
  float s = 0.f;
  for (int j = tid; j < H_; j += 256) s += xr[j];
  red[tid] = s; __syncthreads();
  for (int off = 128; off > 0; off >>= 1) {
    if (tid < off) red[tid] += red[tid + off];
    __syncthreads();
  }
  const float mean = red[0] * (1.f / H_); __syncthreads();
  float vv = 0.f;
  for (int j = tid; j < H_; j += 256) { float d = xr[j] - mean; vv += d * d; }
  red[tid] = vv; __syncthreads();
  for (int off = 128; off > 0; off >>= 1) {
    if (tid < off) red[tid] += red[tid + off];
    __syncthreads();
  }
  const float inv = rsqrtf(red[0] * (1.f / H_) + 1e-5f);
  float* yr = y + (size_t)blockIdx.x * H_;
  for (int j = tid; j < H_; j += 256)
    yr[j] = (xr[j] - mean) * inv * g[j] + bta[j];
}

// ------------------------------------------------------------------
// Monotonic attention recurrence, one block per (b,h):
// alpha_i = p_i * cp_i * cumsum(alpha_{i-1} / cp_i), cp = clipped excl cumprod(1-p)
// ------------------------------------------------------------------
__global__ __launch_bounds__(256) void mono_alpha_kernel(
    const float* __restrict__ p, float* __restrict__ alpha)
{
  __shared__ float red[256];
  const int tid = threadIdx.x;
  const int bh = blockIdx.x;
  const float* pb = p + (size_t)bh * LD_ * LE_;
  float* ab = alpha + (size_t)bh * LD_ * LE_;
  float ap0 = 0.f, ap1 = 0.f, ap2 = 0.f, ap3 = 0.f;
  if (tid == 0) ap0 = 1.f;                         // a0[...,0] = 1
  for (int i = 0; i < LD_; ++i) {
    v4f pv = *(const v4f*)(pb + (size_t)i * LE_ + tid * 4);
    float l0 = 1.f - pv.x;
    float l1 = l0 * (1.f - pv.y);
    float l2 = l1 * (1.f - pv.z);
    float l3 = l2 * (1.f - pv.w);
    red[tid] = l3; __syncthreads();
    for (int off = 1; off < 256; off <<= 1) {       // product scan
      float t = (tid >= off) ? red[tid - off] : 1.f;
      __syncthreads();
      red[tid] *= t;
      __syncthreads();
    }
    float pr = (tid == 0) ? 1.f : red[tid - 1];
    float cp0 = fminf(fmaxf(pr,      1e-10f), 1.f);
    float cp1 = fminf(fmaxf(pr * l0, 1e-10f), 1.f);
    float cp2 = fminf(fmaxf(pr * l1, 1e-10f), 1.f);
    float cp3 = fminf(fmaxf(pr * l2, 1e-10f), 1.f);
    float t0 = ap0 / cp0, t1 = ap1 / cp1, t2 = ap2 / cp2, t3 = ap3 / cp3;
    float s0 = t0, s1 = s0 + t1, s2 = s1 + t2, s3 = s2 + t3;
    __syncthreads();
    red[tid] = s3; __syncthreads();
    for (int off = 1; off < 256; off <<= 1) {       // sum scan
      float t = (tid >= off) ? red[tid - off] : 0.f;
      __syncthreads();
      red[tid] += t;
      __syncthreads();
    }
    float ps = (tid == 0) ? 0.f : red[tid - 1];
    float a0 = pv.x * cp0 * (ps + s0);
    float a1 = pv.y * cp1 * (ps + s1);
    float a2 = pv.z * cp2 * (ps + s2);
    float a3 = pv.w * cp3 * (ps + s3);
    v4f ov; ov.x = a0; ov.y = a1; ov.z = a2; ov.w = a3;
    *(v4f*)(ab + (size_t)i * LE_ + tid * 4) = ov;
    ap0 = a0; ap1 = a1; ap2 = a2; ap3 = a3;
    __syncthreads();
  }
}

// ------------------------------------------------------------------
extern "C" void kernel_launch(void* const* d_in, const int* in_sizes, int n_in,
                              void* d_out, int out_size, void* d_ws, size_t ws_size,
                              hipStream_t stream)
{
  const float* dec   = (const float*)d_in[0];
  const float* dmask = (const float*)d_in[1];
  const float* enc   = (const float*)d_in[2];
  const float* emask = (const float*)d_in[3];
  const float* upper = (const float*)d_in[4];
  const float* lower = (const float*)d_in[5];
  const float* ln1g = (const float*)d_in[6];  const float* ln1b = (const float*)d_in[7];
  const float* ln2g = (const float*)d_in[8];  const float* ln2b = (const float*)d_in[9];
  const float* ln3g = (const float*)d_in[10]; const float* ln3b = (const float*)d_in[11];
  const float* sqkvw = (const float*)d_in[12]; const float* sqkvb = (const float*)d_in[13];
  const float* sow = (const float*)d_in[14];   const float* sob = (const float*)d_in[15];
  const float* cqkvw = (const float*)d_in[16]; const float* cqkvb = (const float*)d_in[17];
  const float* cow = (const float*)d_in[18];   const float* cob = (const float*)d_in[19];
  const float* fw1 = (const float*)d_in[20];   const float* fb1 = (const float*)d_in[21];
  const float* fw2 = (const float*)d_in[22];   const float* fb2 = (const float*)d_in[23];
  const float* flng = (const float*)d_in[24];  const float* flnb = (const float*)d_in[25];

  float* out = (float*)d_out;
  const size_t NX  = (size_t)B_ * LD_ * H_;          // 1,048,576
  const size_t NE  = (size_t)B_ * LE_ * H_;          // 2,097,152
  const size_t NS  = (size_t)B_ * NH_ * LD_ * LD_;   // 8,388,608
  const size_t NF  = (size_t)B_ * LD_ * FI_;         // 4,194,304
  const size_t NAL = (size_t)B_ * NH_ * LD_ * LE_;   // 16,777,216

  float* w  = (float*)d_ws;
  float* x   = w; w += NX;
  float* xn  = w; w += NX;
  float* qs  = w; w += NX;
  float* ks  = w; w += NX;
  float* vs  = w; w += NX;
  float* ctx = w; w += NX;
  float* kc  = w; w += NE;
  float* vc  = w; w += NE;
  float* sc  = w; w += NS;
  float* fh  = w; w += NF;
  // per-layer transposed bf16 weights (reused each layer)
  bf16t* bw = (bf16t*)w;
  bf16t* wq_t  = bw;                       // 3*H*H
  bf16t* wo_t  = wq_t  + (size_t)3*H_*H_;  // H*H
  bf16t* wcq_t = wo_t  + (size_t)H_*H_;    // 3*H*H
  bf16t* wco_t = wcq_t + (size_t)3*H_*H_;  // H*H
  bf16t* wf1_t = wco_t + (size_t)H_*H_;    // H*FI
  bf16t* wf2_t = wf1_t + (size_t)H_*FI_;   // FI*H

  float* out_x = out;
  float* out_a = out + NX;
  float* out_p = out + NX + (size_t)NL_ * NAL;

  (void)hipMemcpyAsync(x, dec, NX * sizeof(float), hipMemcpyDeviceToDevice, stream);

  const int MD = B_ * LD_;   // 1024 rows
  const int ME = B_ * LE_;   // 2048 rows
  dim3 blk(256);

  for (int L = 0; L < NL_; ++L) {
    const float* wqv = sqkvw + (size_t)L * 3 * H_ * H_;
    const float* bqv = sqkvb + (size_t)L * 3 * H_;
    const float* wcv = cqkvw + (size_t)L * 3 * H_ * H_;
    const float* bcv = cqkvb + (size_t)L * 3 * H_;
    // ---- convert+transpose this layer's weights to bf16 [N][K] ----
    for (int sI = 0; sI < 3; ++sI) {
      convert_w_kernel<<<dim3(H_/32, H_/32), blk, 0, stream>>>(wqv + (size_t)sI*H_*H_, wq_t + (size_t)sI*H_*H_, H_, H_);
      convert_w_kernel<<<dim3(H_/32, H_/32), blk, 0, stream>>>(wcv + (size_t)sI*H_*H_, wcq_t + (size_t)sI*H_*H_, H_, H_);
    }
    convert_w_kernel<<<dim3(H_/32, H_/32),  blk, 0, stream>>>(sow + (size_t)L*H_*H_,  wo_t,  H_, H_);
    convert_w_kernel<<<dim3(H_/32, H_/32),  blk, 0, stream>>>(cow + (size_t)L*H_*H_,  wco_t, H_, H_);
    convert_w_kernel<<<dim3(H_/32, FI_/32), blk, 0, stream>>>(fw1 + (size_t)L*H_*FI_, wf1_t, H_, FI_);
    convert_w_kernel<<<dim3(FI_/32, H_/32), blk, 0, stream>>>(fw2 + (size_t)L*FI_*H_, wf2_t, FI_, H_);

    // ---------------- self attention ----------------
    layernorm_kernel<<<MD, blk, 0, stream>>>(x, ln1g + L * H_, ln1b + L * H_, xn);
    gemm_kernel<false,false><<<dim3(MD/128, H_/128), blk, 0, stream>>>(xn, wq_t,                   bqv,        nullptr, qs, MD, H_, H_);
    gemm_kernel<false,false><<<dim3(MD/128, H_/128), blk, 0, stream>>>(xn, wq_t + (size_t)H_*H_,   bqv + H_,   nullptr, ks, MD, H_, H_);
    gemm_kernel<false,false><<<dim3(MD/128, H_/128), blk, 0, stream>>>(xn, wq_t + (size_t)2*H_*H_, bqv + 2*H_, nullptr, vs, MD, H_, H_);
    self_scores_kernel<<<dim3(LD_/64, LD_/128, B_*NH_), blk, 0, stream>>>(qs, ks, dmask, sc);
    softmax_kernel<<<B_*NH_*LD_, blk, 0, stream>>>(sc, LD_);
    attn_context_kernel<<<dim3(LD_/128, B_*NH_), blk, 0, stream>>>(sc, vs, ctx, LD_);
    gemm_kernel<true,false><<<dim3(MD/128, H_/128), blk, 0, stream>>>(ctx, wo_t, sob + L*H_, x, x, MD, H_, H_);

    // ------------- monotonic cross attention -------------
    layernorm_kernel<<<MD, blk, 0, stream>>>(x, ln2g + L * H_, ln2b + L * H_, xn);
    gemm_kernel<false,false><<<dim3(MD/128, H_/128), blk, 0, stream>>>(xn,  wcq_t,                   bcv,        nullptr, qs, MD, H_, H_);
    gemm_kernel<false,false><<<dim3(ME/128, H_/128), blk, 0, stream>>>(enc, wcq_t + (size_t)H_*H_,   bcv + H_,   nullptr, kc, ME, H_, H_);
    gemm_kernel<false,false><<<dim3(ME/128, H_/128), blk, 0, stream>>>(enc, wcq_t + (size_t)2*H_*H_, bcv + 2*H_, nullptr, vc, ME, H_, H_);
    float* pL = out_p + (size_t)L * NAL;
    float* aL = out_a + (size_t)L * NAL;
    cross_p_kernel<<<dim3(LD_/64, LE_/128, B_*NH_), blk, 0, stream>>>(qs, kc, emask, upper, lower, pL);
    mono_alpha_kernel<<<B_*NH_, blk, 0, stream>>>(pL, aL);
    attn_context_kernel<<<dim3(LD_/128, B_*NH_), blk, 0, stream>>>(aL, vc, ctx, LE_);
    gemm_kernel<true,false><<<dim3(MD/128, H_/128), blk, 0, stream>>>(ctx, wco_t, cob + L*H_, x, x, MD, H_, H_);

    // ---------------- FFN ----------------
    layernorm_kernel<<<MD, blk, 0, stream>>>(x, ln3g + L * H_, ln3b + L * H_, xn);
    gemm_kernel<false,true><<<dim3(MD/128, FI_/128), blk, 0, stream>>>(xn, wf1_t, fb1 + L*FI_, nullptr, fh, MD, FI_, H_);
    gemm_kernel<true,false><<<dim3(MD/128, H_/128),  blk, 0, stream>>>(fh, wf2_t, fb2 + L*H_,  x,      x,  MD, H_, FI_);
  }
  layernorm_kernel<<<MD, blk, 0, stream>>>(x, flng, flnb, out_x);
}
